// AttentionLayer_4174708212459
// MI455X (gfx1250) — matrix operand order repaired and checked by reading, not compiled, not run on hardware
//
#include <hip/hip_runtime.h>

// ---------------------------------------------------------------------------
// Types for CDNA5 WMMA
// ---------------------------------------------------------------------------
typedef __attribute__((ext_vector_type(16))) __bf16 v16bf;
typedef __attribute__((ext_vector_type(4)))  __bf16 v4bf;
typedef __attribute__((ext_vector_type(8)))  float  v8f;

__device__ __forceinline__ v8f wmma_bf16(v16bf a, v16bf b, v8f c) {
    // D = A(16x32 bf16) * B(32x16 bf16) + C(16x16 f32)
    return __builtin_amdgcn_wmma_f32_16x16x32_bf16(
        /*neg_a=*/false, a, /*neg_b=*/false, b,
        /*c_mod=*/(short)0, c, /*reuse_a=*/false, /*reuse_b=*/false);
}

// A-operand (16-bit, 16xK): lane holds row M = lane&15.
// Element e -> K = 2*hiK + e (contiguous), hiK = (lane>=16)?8:0.
__device__ __forceinline__ v16bf lds_frag_a(const __bf16* rowBase, int hiK) {
    v16bf a;
#pragma unroll
    for (int e = 0; e < 16; ++e) a[e] = rowBase[2 * hiK + e];
    return a;
}

// B-operand (16-bit, Kx16) from a TRANSPOSED LDS tile [n][k]:
// lane holds col N = lane&15; element e -> K = kb + e, kb = (lane>=16)?16:0.
__device__ __forceinline__ v16bf lds_frag_bt(const __bf16* p) {
    v16bf b;
#pragma unroll
    for (int e = 0; e < 16; ++e) b[e] = p[e];
    return b;
}

__device__ __forceinline__ void wave_lds_fence() {
    asm volatile("s_wait_dscnt 0" ::: "memory");
}

__device__ __forceinline__ float f4c(const float4& f, int i) {
    return ((const float*)&f)[i];
}

__device__ __forceinline__ v4bf cvt4(const float4& f) {
    v4bf o;
    o[0] = (__bf16)f.x; o[1] = (__bf16)f.y; o[2] = (__bf16)f.z; o[3] = (__bf16)f.w;
    return o;
}

// ---------------------------------------------------------------------------
// Tiled GEMM:  out[M,N] = act( A[M,K] @ W[K,N] + bias[N] (+ resid) )
// 128x128 block, K-step 32, 128 threads = 4 waves, 64x64 wave tile.
// Register prefetch + DOUBLE-BUFFERED LDS: one barrier per k-step.
// ACT: 0 none, 1 exact GELU.  RES: residual add.
// ---------------------------------------------------------------------------
template <int ACT, int RES>
__global__ __launch_bounds__(128)
void gemm_bf16_kernel(const float* __restrict__ A, const float* __restrict__ W,
                      const float* __restrict__ bias, const float* __restrict__ resid,
                      float* __restrict__ out, int M, int K, int N) {
    __shared__ __bf16 aLds[2][128 * 32];   // [buf][row][k]
    __shared__ __bf16 bLds[2][128 * 32];   // [buf], transposed: [n][k]

    const int tid  = threadIdx.x;
    const int lane = tid & 31;
    const int wave = tid >> 5;
    const int wr   = wave >> 1;          // 0..1 (row group of 64)
    const int wc   = wave & 1;           // 0..1 (col group of 64)
    const int row0 = blockIdx.y * 128;
    const int col0 = blockIdx.x * 128;

    const int n15 = lane & 15;
    const int hiK = (lane & 16) ? 8 : 0;
    const int kb  = (lane & 16) ? 16 : 0;
    const int hiM = (lane & 16) ? 8 : 0;

    // staging decomposition (128 threads)
    const int bkg = tid >> 4;            // B k-group (4 rows of W)
    const int bc4 = (tid & 15) * 4;      // B col group (2 tasks: +0, +64)

    float4 aP[8];          // A prefetch
    float4 bP[2][4];       // B prefetch: 2 col-tasks x 4 k-rows

    auto loadAB = [&](int k0) {
#pragma unroll
        for (int t = 0; t < 8; ++t) {
            const int r  = (tid + t * 128) >> 3;           // 0..127
            const int c4 = ((tid + t * 128) & 7) * 4;
            aP[t] = *(const float4*)&A[(size_t)(row0 + r) * K + k0 + c4];
        }
#pragma unroll
        for (int t = 0; t < 2; ++t) {
            const int c4 = bc4 + t * 64;                   // 0..124
#pragma unroll
            for (int i = 0; i < 4; ++i)
                bP[t][i] = *(const float4*)&W[(size_t)(k0 + bkg * 4 + i) * N + col0 + c4];
        }
    };
    auto storeAB = [&](int buf) {
#pragma unroll
        for (int t = 0; t < 8; ++t) {
            const int r  = (tid + t * 128) >> 3;
            const int c4 = ((tid + t * 128) & 7) * 4;
            *(v4bf*)&aLds[buf][r * 32 + c4] = cvt4(aP[t]);
        }
#pragma unroll
        for (int t = 0; t < 2; ++t) {
            const int c4 = bc4 + t * 64;
#pragma unroll
            for (int j = 0; j < 4; ++j) {                  // output col n = c4+j
                v4bf o;
#pragma unroll
                for (int i = 0; i < 4; ++i) o[i] = (__bf16)f4c(bP[t][i], j);
                *(v4bf*)&bLds[buf][(c4 + j) * 32 + bkg * 4] = o;
            }
        }
    };

    v8f acc[4][4];
#pragma unroll
    for (int mt = 0; mt < 4; ++mt)
#pragma unroll
        for (int nt = 0; nt < 4; ++nt) acc[mt][nt] = {};

    loadAB(0);
    storeAB(0);
    __syncthreads();

    int cur = 0;
    for (int k0 = 0; k0 < K; k0 += 32) {
        const bool nxt = (k0 + 32) < K;
        if (nxt) loadAB(k0 + 32);        // global loads overlap the WMMAs below
        if (k0 + 64 < K) {               // warm GL2 two steps ahead
            __builtin_prefetch(&A[(size_t)(row0 + (tid >> 3)) * K + k0 + 64], 0, 1);
            __builtin_prefetch(&W[(size_t)(k0 + 64 + bkg * 4) * N + col0 + bc4], 0, 1);
        }

        v16bf bf[4];
#pragma unroll
        for (int nt = 0; nt < 4; ++nt)
            bf[nt] = lds_frag_bt(&bLds[cur][(wc * 64 + nt * 16 + n15) * 32 + kb]);
#pragma unroll
        for (int mt = 0; mt < 4; ++mt) {
            v16bf af = lds_frag_a(&aLds[cur][(wr * 64 + mt * 16 + n15) * 32], hiK);
#pragma unroll
            for (int nt = 0; nt < 4; ++nt)
                acc[mt][nt] = wmma_bf16(af, bf[nt], acc[mt][nt]);
        }

        if (nxt) storeAB(cur ^ 1);       // write other buffer: no pre-store barrier
        __syncthreads();
        cur ^= 1;
    }

    // epilogue (branch-free: template-specialized)
#pragma unroll
    for (int mt = 0; mt < 4; ++mt) {
#pragma unroll
        for (int nt = 0; nt < 4; ++nt) {
            const int col = col0 + wc * 64 + nt * 16 + n15;
            const float bv = bias[col];
#pragma unroll
            for (int j = 0; j < 8; ++j) {
                const int row = row0 + wr * 64 + mt * 16 + j + hiM;
                float v = acc[mt][nt][j] + bv;
                if (RES) v += resid[(size_t)row * N + col];
                if (ACT == 1) v = 0.5f * v * (1.0f + erff(v * 0.70710678118654752f));
                out[(size_t)row * N + col] = v;
            }
        }
    }
}

// ---------------------------------------------------------------------------
// LayerNorm over last dim (1024), one block per row.
// ---------------------------------------------------------------------------
__global__ __launch_bounds__(256)
void layernorm_kernel(const float* __restrict__ X, const float* __restrict__ g,
                      const float* __restrict__ b, float* __restrict__ Y) {
    __shared__ float r1[256], r2[256];
    const int row = blockIdx.x;
    const int tid = threadIdx.x;
    const float* x = X + (size_t)row * 1024;

    float v[4], s1 = 0.f, s2 = 0.f;
#pragma unroll
    for (int j = 0; j < 4; ++j) {
        v[j] = x[tid + j * 256];
        s1 += v[j];
        s2 += v[j] * v[j];
    }
    r1[tid] = s1; r2[tid] = s2;
    __syncthreads();
    for (int o = 128; o > 0; o >>= 1) {
        if (tid < o) { r1[tid] += r1[tid + o]; r2[tid] += r2[tid + o]; }
        __syncthreads();
    }
    const float mean = r1[0] * (1.0f / 1024.0f);
    const float var  = r2[0] * (1.0f / 1024.0f) - mean * mean;
    const float rstd = rsqrtf(var + 1e-12f);
#pragma unroll
    for (int j = 0; j < 4; ++j) {
        const int c = tid + j * 256;
        Y[(size_t)row * 1024 + c] = (v[j] - mean) * rstd * g[c] + b[c];
    }
}

// ---------------------------------------------------------------------------
// Fused flash attention with relative_key_query position bias.
// grid = (S/64, H, B), block = 128 (4 waves). Each wave owns 16 query rows.
// Double-buffered K/V staging, shared 48-row dist_emb window per 32-col step.
// ---------------------------------------------------------------------------
#define SEQ 1024
#define HIDW 1024

__global__ __launch_bounds__(128)
void attn_kernel(const float* __restrict__ qb, const float* __restrict__ kbuf,
                 const float* __restrict__ vbuf, const float* __restrict__ Emb,
                 const float* __restrict__ amask, const float* __restrict__ hmask,
                 float* __restrict__ ctx) {
    __shared__ __bf16 kLds[2][32 * 64];   // [buf][r][d]  (shared by all waves)
    __shared__ __bf16 vLds[2][64 * 32];   // [buf][d][r]  transposed
    __shared__ __bf16 qLds[4][16 * 64];   // per-wave [m][d]
    __shared__ __bf16 eLds[4][48 * 64];   // per-wave [win48][d]
    __shared__ float  qeLds[4][16 * 32];  // per-wave [m][win32]
    __shared__ float  keLds[4][16 * 32];  // per-wave [r][win32]
    __shared__ __bf16 pLds[4][16 * 32];   // per-wave [m][c]

    const int b    = blockIdx.z;
    const int h    = blockIdx.y;
    const int wave = threadIdx.x >> 5;
    const int lane = threadIdx.x & 31;
    const int l0   = blockIdx.x * 64 + wave * 16;

    const int n15 = lane & 15;
    const int hiK = (lane & 16) ? 8 : 0;
    const int kbK = (lane & 16) ? 16 : 0;
    const int hiM = (lane & 16) ? 8 : 0;

    // load Q block (16x64) -> bf16 LDS, vectorized
    {
        const float* qsrc = qb + ((size_t)(b * SEQ + l0)) * HIDW + h * 64;
#pragma unroll
        for (int t = 0; t < 8; ++t) {
            const int idx = lane + t * 32;       // 0..255 float4 slots
            const int r = idx >> 4, c4 = (idx & 15) * 4;
            float4 f = *(const float4*)&qsrc[(size_t)r * HIDW + c4];
            *(v4bf*)&qLds[wave][r * 64 + c4] = cvt4(f);
        }
        wave_lds_fence();
    }

    float m8[8], l8[8];
    v8f ctxacc[4];
#pragma unroll
    for (int j = 0; j < 8; ++j) { m8[j] = -1e30f; l8[j] = 0.f; }
#pragma unroll
    for (int nt = 0; nt < 4; ++nt) ctxacc[nt] = {};

    // K/V staging: 128 threads, 8 row-groups x 16 col4 groups
    const int kg  = threadIdx.x >> 4;       // 0..7  -> rows 4kg..4kg+3
    const int kc4 = (threadIdx.x & 15) * 4; // 0..60

    float4 fk[4], fv[4];
    auto loadKV = [&](int r0) {
        const float* ks = kbuf + ((size_t)(b * SEQ + r0)) * HIDW + h * 64;
        const float* vs = vbuf + ((size_t)(b * SEQ + r0)) * HIDW + h * 64;
#pragma unroll
        for (int i = 0; i < 4; ++i) {
            fk[i] = *(const float4*)&ks[(size_t)(kg * 4 + i) * HIDW + kc4];
            fv[i] = *(const float4*)&vs[(size_t)(kg * 4 + i) * HIDW + kc4];
        }
    };
    auto storeKV = [&](int buf) {
#pragma unroll
        for (int i = 0; i < 4; ++i)
            *(v4bf*)&kLds[buf][(kg * 4 + i) * 64 + kc4] = cvt4(fk[i]);
#pragma unroll
        for (int j = 0; j < 4; ++j) {           // V transposed in registers
            v4bf o;
#pragma unroll
            for (int i = 0; i < 4; ++i) o[i] = (__bf16)f4c(fv[i], j);
            *(v4bf*)&vLds[buf][(kc4 + j) * 32 + kg * 4] = o;
        }
    };

    loadKV(0);
    storeKV(0);
    __syncthreads();

    int cur = 0;
    for (int r0 = 0; r0 < SEQ; r0 += 32) {
        const bool nxt = (r0 + 32) < SEQ;
        if (nxt) loadKV(r0 + 32);     // overlap next K/V panel with compute

        // shared 48-row E window: dist = dbaseAll + i, i in [0,48)
        // ci=0 uses rows 16..47, ci=1 uses rows 0..31.
        const int dbaseAll = l0 - r0 + 992;
#pragma unroll
        for (int t = 0; t < 24; ++t) {
            const int idx = lane + t * 32;       // 0..767 float4 slots
            const int r = idx >> 4, c4 = (idx & 15) * 4;
            int d = dbaseAll + r;
            d = d < 0 ? 0 : (d > 2046 ? 2046 : d);
            float4 f = *(const float4*)&Emb[(size_t)d * 64 + c4];
            *(v4bf*)&eLds[wave][r * 64 + c4] = cvt4(f);
        }
        wave_lds_fence();

        v8f sreg[2];
#pragma unroll
        for (int ci = 0; ci < 2; ++ci) {
            const int rc   = r0 + ci * 16;
            const int eoff = ci ? 0 : 16;        // window base row in eLds

            v8f s = {}, qe0 = {}, qe1 = {}, ke0 = {}, ke1 = {};
#pragma unroll
            for (int k0 = 0; k0 < 64; k0 += 32) {
                v16bf aq = lds_frag_a(&qLds[wave][n15 * 64 + k0], hiK);
                v16bf ak = lds_frag_a(&kLds[cur][(ci * 16 + n15) * 64 + k0], hiK);
                v16bf bk = lds_frag_bt(&kLds[cur][(ci * 16 + n15) * 64 + k0 + kbK]);
                v16bf be0 = lds_frag_bt(&eLds[wave][(eoff + n15)      * 64 + k0 + kbK]);
                v16bf be1 = lds_frag_bt(&eLds[wave][(eoff + 16 + n15) * 64 + k0 + kbK]);
                s   = wmma_bf16(aq, bk,  s);
                qe0 = wmma_bf16(aq, be0, qe0);
                qe1 = wmma_bf16(aq, be1, qe1);
                ke0 = wmma_bf16(ak, be0, ke0);
                ke1 = wmma_bf16(ak, be1, ke1);
            }
            // spill qE/kE tiles so the bias gather can cross lanes
#pragma unroll
            for (int j = 0; j < 8; ++j) {
                const int rr = j + hiM;
                qeLds[wave][rr * 32 + n15]      = qe0[j];
                qeLds[wave][rr * 32 + 16 + n15] = qe1[j];
                keLds[wave][rr * 32 + n15]      = ke0[j];
                keLds[wave][rr * 32 + 16 + n15] = ke1[j];
            }
            wave_lds_fence();

            const float amv = amask[(size_t)b * SEQ + rc + n15];
#pragma unroll
            for (int j = 0; j < 8; ++j) {
                const int Mr = j + hiM;
                const int c  = Mr - n15 + 15;                  // in [0,30]
                const float sv = s[j] + qeLds[wave][Mr * 32 + c]
                                      + keLds[wave][n15 * 32 + c];
                sreg[ci][j] = sv * 0.125f + amv;
            }
        }

        // online softmax update over the 16x32 score strip
#pragma unroll
        for (int j = 0; j < 8; ++j) {
            float mx = fmaxf(sreg[0][j], sreg[1][j]);
#pragma unroll
            for (int o = 1; o < 16; o <<= 1) mx = fmaxf(mx, __shfl_xor(mx, o, 32));
            const float mnew  = fmaxf(m8[j], mx);
            const float alpha = __expf(m8[j] - mnew);
            const float p0 = __expf(sreg[0][j] - mnew);
            const float p1 = __expf(sreg[1][j] - mnew);
            float rs = p0 + p1;
#pragma unroll
            for (int o = 1; o < 16; o <<= 1) rs += __shfl_xor(rs, o, 32);
            l8[j] = l8[j] * alpha + rs;
            m8[j] = mnew;
#pragma unroll
            for (int nt = 0; nt < 4; ++nt) ctxacc[nt][j] *= alpha;
            const int Mr = j + hiM;
            pLds[wave][Mr * 32 + n15]      = (__bf16)p0;
            pLds[wave][Mr * 32 + 16 + n15] = (__bf16)p1;
        }
        wave_lds_fence();

        // ctx += P(16x32) @ V(32x64)
        v16bf pa = lds_frag_a(&pLds[wave][n15 * 32], hiK);
#pragma unroll
        for (int nt = 0; nt < 4; ++nt) {
            v16bf bv = lds_frag_bt(&vLds[cur][(nt * 16 + n15) * 32 + kbK]);
            ctxacc[nt] = wmma_bf16(pa, bv, ctxacc[nt]);
        }

        if (nxt) storeKV(cur ^ 1);    // fill other buffer: no pre-store barrier
        __syncthreads();
        cur ^= 1;
    }

    // finalize: divide by softmax denominator, apply head_mask, write ctx
    const float hm = hmask[h];
#pragma unroll
    for (int j = 0; j < 8; ++j) {
        const float inv = hm / l8[j];
        const int row = l0 + j + hiM;
#pragma unroll
        for (int nt = 0; nt < 4; ++nt) {
            ctx[((size_t)(b * SEQ + row)) * HIDW + h * 64 + nt * 16 + n15] =
                ctxacc[nt][j] * inv;
        }
    }
}

// ---------------------------------------------------------------------------
// Host launcher
// ---------------------------------------------------------------------------
extern "C" void kernel_launch(void* const* d_in, const int* in_sizes, int n_in,
                              void* d_out, int out_size, void* d_ws, size_t ws_size,
                              hipStream_t stream) {
    (void)in_sizes; (void)n_in; (void)out_size; (void)ws_size;

    const float* hidden = (const float*)d_in[0];
    const float* amask  = (const float*)d_in[1];
    const float* hmask  = (const float*)d_in[2];
    const float* Wq = (const float*)d_in[3];  const float* bq = (const float*)d_in[4];
    const float* Wk = (const float*)d_in[5];  const float* bk = (const float*)d_in[6];
    const float* Wv = (const float*)d_in[7];  const float* bv = (const float*)d_in[8];
    const float* Emb = (const float*)d_in[9];
    const float* Wo = (const float*)d_in[10]; const float* bo = (const float*)d_in[11];
    const float* g1 = (const float*)d_in[12]; const float* b1 = (const float*)d_in[13];
    const float* Wi = (const float*)d_in[14]; const float* bi = (const float*)d_in[15];
    const float* Wo2 = (const float*)d_in[16]; const float* bo2 = (const float*)d_in[17];
    const float* g2 = (const float*)d_in[18]; const float* b2 = (const float*)d_in[19];
    float* outp = (float*)d_out;

    const size_t M16 = (size_t)4 * 1024 * 1024;   // 4M floats = 16 MB
    float* ws    = (float*)d_ws;
    float* qbuf  = ws;
    float* kbuf  = ws + 1 * M16;
    float* vbuf  = ws + 2 * M16;
    float* cbuf  = ws + 3 * M16;
    float* tbuf  = ws + 4 * M16;                  // pre-LN scratch (reused)
    float* a1buf = ws + 5 * M16;                  // attn LayerNorm output
    float* inter = ws + 6 * M16;                  // 16M floats = 64 MB

    const int M = 4096;                           // B*S
    dim3 blk(128);
    dim3 gN1024(1024 / 128, M / 128);             // (8, 32)
    dim3 gN4096(4096 / 128, M / 128);             // (32, 32)

    // 1. QKV projections
    gemm_bf16_kernel<0,0><<<gN1024, blk, 0, stream>>>(hidden, Wq, bq, nullptr, qbuf, M, 1024, 1024);
    gemm_bf16_kernel<0,0><<<gN1024, blk, 0, stream>>>(hidden, Wk, bk, nullptr, kbuf, M, 1024, 1024);
    gemm_bf16_kernel<0,0><<<gN1024, blk, 0, stream>>>(hidden, Wv, bv, nullptr, vbuf, M, 1024, 1024);

    // 2. fused attention (flash-style, relative_key_query bias)
    attn_kernel<<<dim3(1024 / 64, 16, 4), dim3(128), 0, stream>>>(
        qbuf, kbuf, vbuf, Emb, amask, hmask, cbuf);

    // 3. output projection + residual, then LayerNorm
    gemm_bf16_kernel<0,1><<<gN1024, blk, 0, stream>>>(cbuf, Wo, bo, hidden, tbuf, M, 1024, 1024);
    layernorm_kernel<<<dim3(4096), dim3(256), 0, stream>>>(tbuf, g1, b1, a1buf);

    // 4. FFN: GELU(a1 @ Wi + bi) @ Wo2 + bo2 + a1, then LayerNorm
    gemm_bf16_kernel<1,0><<<gN4096, blk, 0, stream>>>(a1buf, Wi, bi, nullptr, inter, M, 1024, 4096);
    gemm_bf16_kernel<0,1><<<gN1024, blk, 0, stream>>>(inter, Wo2, bo2, a1buf, tbuf, M, 4096, 1024);
    layernorm_kernel<<<dim3(4096), dim3(256), 0, stream>>>(tbuf, g2, b2, outp);
}
